// MLP_INT8_14748917694594
// MI455X (gfx1250) — 8-bit matrix operand order repaired and checked by reading, compiled, not run on hardware
//
#include <hip/hip_runtime.h>

typedef __attribute__((ext_vector_type(8))) int   v8i;
typedef __attribute__((ext_vector_type(4))) float v4f;

#define NWAVES  8
#define HSTR    48

__device__ __forceinline__ float amax4(v4f x) {
  return fmaxf(fmaxf(fabsf(x.x), fabsf(x.y)), fmaxf(fabsf(x.z), fabsf(x.w)));
}

__device__ __forceinline__ unsigned pack4(v4f x, float inv) {
  int q0 = __float2int_rn(x.x * inv);
  int q1 = __float2int_rn(x.y * inv);
  int q2 = __float2int_rn(x.z * inv);
  int q3 = __float2int_rn(x.w * inv);
  return (unsigned)(q0 & 0xFF) | ((unsigned)(q1 & 0xFF) << 8) |
         ((unsigned)(q2 & 0xFF) << 16) | ((unsigned)(q3 & 0xFF) << 24);
}

__device__ __forceinline__ int a_koff(int v, int g) {
  return 32*(v >> 2) + 16*((v >> 1) & 1) + 4*(v & 1) + 8*g;
}
__device__ __forceinline__ int b_koff(int v, int g) {
  return 32*(v >> 2) + 16*((v >> 1) & 1) + 4*(v & 1) + 8*g;
}

__global__ void __launch_bounds__(256)
mlp_int8_wmma_kernel(const float* __restrict__ x,
                     const float* __restrict__ W1, const float* __restrict__ b1,
                     const float* __restrict__ W2, const float* __restrict__ b2,
                     const float* __restrict__ W3, const float* __restrict__ b3,
                     const float* __restrict__ W4, const float* __restrict__ b4,
                     float* __restrict__ out, int Brows, int tiles)
{
  __shared__ unsigned s_wq[4096];
  __shared__ float s_sw[128], s_invsw[128], s_bias[128];
  __shared__ float s_h[NWAVES][16 * HSTR];
  __shared__ float s_sx[NWAVES][16];

  const int tid = threadIdx.x;

  {
    const float* Wr = nullptr; const float* bb = nullptr;
    int len = 0, sof = 0, n = -1;
    if (tid < 40)                      { n = tid;       Wr = W1 + n*160; len = 160; sof = 0;   bb = b1; }
    else if (tid >= 64  && tid < 104)  { n = tid - 64;  Wr = W2 + n*40;  len = 40;  sof = 40;  bb = b2; }
    else if (tid >= 128 && tid < 168)  { n = tid - 128; Wr = W3 + n*40;  len = 40;  sof = 80;  bb = b3; }
    else if (tid >= 192 && tid < 194)  { n = tid - 192; Wr = W4 + n*40;  len = 40;  sof = 120; bb = b4; }
    if (n >= 0) {
      float am = 0.f;
      for (int k = 0; k < len; ++k) am = fmaxf(am, fabsf(Wr[k]));
      float s = fmaxf(am * (1.0f/127.0f), 1e-8f);
      s_sw[sof + n] = s; s_invsw[sof + n] = 1.0f / s; s_bias[sof + n] = bb[n];
    }
  }
  __syncthreads();

  for (int f = tid; f < 4096; f += 256) {
    const float* W; int KB, INl, OUTl, sof, local;
    if (f < 2304)      { W = W1; KB = 3; INl = 160; OUTl = 40; sof = 0;   local = f;        }
    else if (f < 3072) { W = W2; KB = 1; INl = 40;  OUTl = 40; sof = 40;  local = f - 2304; }
    else if (f < 3840) { W = W3; KB = 1; INl = 40;  OUTl = 40; sof = 80;  local = f - 3072; }
    else               { W = W4; KB = 1; INl = 40;  OUTl = 2;  sof = 120; local = f - 3840; }
    int chunk = local >> 8, idx = local & 255;
    int nt = chunk / KB, kb = chunk % KB;
    int lane2 = idx >> 3, v = idx & 7;
    int n = nt*16 + (lane2 & 15), g2 = lane2 >> 4;
    int k0 = kb*64 + b_koff(v, g2);
    unsigned d = 0;
    if (n < OUTl) {
      float inv = s_invsw[sof + n];
      #pragma unroll
      for (int j = 0; j < 4; ++j) {
        int k = k0 + j;
        int q = (k < INl) ? __float2int_rn(W[n*INl + k] * inv) : 0;
        d |= ((unsigned)(q & 0xFF)) << (8*j);
      }
    }
    s_wq[f] = d;
  }
  __syncthreads();

  const int lane = tid & 31, wave = tid >> 5;
  const int r = lane & 15, g = lane >> 4;
  float* hbuf  = &s_h[wave][0];
  float* sxbuf = &s_sx[wave][0];
  const v8i zero = {};

  for (int t = blockIdx.x*NWAVES + wave; t < tiles; t += gridDim.x*NWAVES) {
    int tn = t + gridDim.x*NWAVES;
    if (tn < tiles) __builtin_prefetch(x + (size_t)(tn*16 + r)*160, 0, 3);

    int row = t*16 + r;
    const float* xr = x + (size_t)min(row, Brows - 1) * 160;
    v4f xa[3][8];
    #pragma unroll
    for (int kb = 0; kb < 3; ++kb)
      #pragma unroll
      for (int v = 0; v < 8; ++v) {
        int k0 = kb*64 + a_koff(v, g);
        v4f tmp = {};
        if (k0 + 4 <= 160) tmp = *(const v4f*)(xr + k0);
        xa[kb][v] = tmp;
      }
    float am = 0.f;
    #pragma unroll
    for (int kb = 0; kb < 3; ++kb)
      #pragma unroll
      for (int v = 0; v < 8; ++v) am = fmaxf(am, amax4(xa[kb][v]));
    am = fmaxf(am, __shfl_xor(am, 16, 32));
    float s1 = fmaxf(am * (1.0f/127.0f), 1e-8f), inv1 = 1.0f/s1;
    if (g == 0) sxbuf[r] = s1;
    v8i a1[3];
    #pragma unroll
    for (int kb = 0; kb < 3; ++kb)
      #pragma unroll
      for (int v = 0; v < 8; ++v) a1[kb][v] = (int)pack4(xa[kb][v], inv1);

    v8i acc[3];
    #pragma unroll
    for (int nt = 0; nt < 3; ++nt) {
      acc[nt] = zero;
      #pragma unroll
      for (int kb = 0; kb < 3; ++kb) {
        v8i bm = *(const v8i*)(s_wq + (nt*3 + kb)*256 + lane*8);
        acc[nt] = __builtin_amdgcn_wmma_i32_16x16x64_iu8(true, a1[kb], true, bm, acc[nt], false, false);
      }
    }
    #pragma unroll
    for (int nt = 0; nt < 3; ++nt) {
      int n = nt*16 + r;
      if (n < 40) {
        float swv = s_sw[n], bb = s_bias[n];
        #pragma unroll
        for (int j = 0; j < 8; ++j) {
          int m = g*8 + j;
          float val = (float)acc[nt][j] * (sxbuf[m] * swv) + bb;
          hbuf[m*HSTR + n] = tanhf(val);
        }
      }
    }

    #pragma unroll
    for (int layer = 0; layer < 2; ++layer) {
      const unsigned* wq = s_wq + (layer == 0 ? 2304 : 3072);
      const float* swp   = s_sw   + (layer == 0 ? 40 : 80);
      const float* bsp   = s_bias + (layer == 0 ? 40 : 80);
      v4f hx[10];
      #pragma unroll
      for (int i = 0; i < 10; ++i) hx[i] = *(const v4f*)(hbuf + r*HSTR + i*4);
      float hm = 0.f;
      #pragma unroll
      for (int i = 0; i < 10; ++i) hm = fmaxf(hm, amax4(hx[i]));
      float s = fmaxf(hm * (1.0f/127.0f), 1e-8f), inv = 1.0f/s;
      if (g == 0) sxbuf[r] = s;
      v8i a;
      #pragma unroll
      for (int v = 0; v < 8; ++v) {
        int k0 = a_koff(v, g);
        a[v] = (k0 < 40) ? (int)pack4(hx[k0 >> 2], inv) : 0;
      }
      v8i hacc[3];
      #pragma unroll
      for (int nt = 0; nt < 3; ++nt) {
        v8i bm = *(const v8i*)(wq + nt*256 + lane*8);
        hacc[nt] = __builtin_amdgcn_wmma_i32_16x16x64_iu8(true, a, true, bm, zero, false, false);
      }
      #pragma unroll
      for (int nt = 0; nt < 3; ++nt) {
        int n = nt*16 + r;
        if (n < 40) {
          float swv = swp[n], bb = bsp[n];
          #pragma unroll
          for (int j = 0; j < 8; ++j) {
            int m = g*8 + j;
            float val = (float)hacc[nt][j] * (sxbuf[m] * swv) + bb;
            hbuf[m*HSTR + n] = (val > 0.f) ? val : expm1f(val);
          }
        }
      }
    }

    {
      v4f hx[10];
      #pragma unroll
      for (int i = 0; i < 10; ++i) hx[i] = *(const v4f*)(hbuf + r*HSTR + i*4);
      float hm = 0.f;
      #pragma unroll
      for (int i = 0; i < 10; ++i) hm = fmaxf(hm, amax4(hx[i]));
      float s = fmaxf(hm * (1.0f/127.0f), 1e-8f), inv = 1.0f/s;
      if (g == 0) sxbuf[r] = s;
      v8i a;
      #pragma unroll
      for (int v = 0; v < 8; ++v) {
        int k0 = a_koff(v, g);
        a[v] = (k0 < 40) ? (int)pack4(hx[k0 >> 2], inv) : 0;
      }
      v8i bm = *(const v8i*)(s_wq + 3840 + lane*8);
      v8i oacc = __builtin_amdgcn_wmma_i32_16x16x64_iu8(true, a, true, bm, zero, false, false);
      int n = r;
      if (n < 2) {
        float swv = s_sw[120 + n], bb = s_bias[120 + n];
        #pragma unroll
        for (int j = 0; j < 8; ++j) {
          int m = g*8 + j;
          int ro = t*16 + m;
          float val = (float)oacc[j] * (sxbuf[m] * swv) + bb;
          float sp = fmaxf(val, 0.f) + log1pf(expf(-fabsf(val)));
          if (ro < Brows) out[(size_t)ro*2 + n] = sp;
        }
      }
    }
  }
}

extern "C" void kernel_launch(void* const* d_in, const int* in_sizes, int n_in,
                              void* d_out, int out_size, void* d_ws, size_t ws_size,
                              hipStream_t stream) {
  (void)n_in; (void)out_size; (void)d_ws; (void)ws_size;
  const float* x  = (const float*)d_in[0];
  const float* W1 = (const float*)d_in[1];
  const float* b1 = (const float*)d_in[2];
  const float* W2 = (const float*)d_in[3];
  const float* b2 = (const float*)d_in[4];
  const float* W3 = (const float*)d_in[5];
  const float* b3 = (const float*)d_in[6];
  const float* W4 = (const float*)d_in[7];
  const float* b4 = (const float*)d_in[8];
  float* out = (float*)d_out;

  int Brows = in_sizes[0] / 160;
  int tiles = (Brows + 15) / 16;
  int nblk  = 1024;
  if (tiles < nblk * NWAVES) nblk = (tiles + NWAVES - 1) / NWAVES;
  if (nblk < 1) nblk = 1;

  mlp_int8_wmma_kernel<<<nblk, 256, 0, stream>>>(x, W1, b1, W2, b2, W3, b3, W4, b4,
                                                 out, Brows, tiles);
}
